// SelectiveScanFFN_89799176225253
// MI455X (gfx1250) — matrix operand order repaired and checked by reading, not compiled, not run on hardware
//
#include <hip/hip_runtime.h>
#include <hip/hip_bf16.h>
#include <math.h>

// Problem sizes (fixed by the reference)
#define BB    4
#define SS    4096
#define DD    2048
#define NSCAN 128
#define NST   16
#define MM    (BB * SS)   // 16384 token rows

#define KC     128        // K-chunk staged in LDS per block iteration
#define LDSROW 136        // padded LDS row stride (bf16 elems): 272B = 68 dwords
#define MATELEMS (128 * LDSROW)        // one staged weight matrix chunk
#define BUFELEMS (2 * MATELEMS)        // Wi + Wg chunk
#define SHBYTES  (2 * BUFELEMS * 2)    // double-buffered, bf16 -> 139,264 B

// Use CDNA5 async global->LDS DMA for weight staging (ASYNCcnt path).
#define STAGE_ASYNC 1

typedef __attribute__((ext_vector_type(16))) __bf16 v16bf;
typedef __attribute__((ext_vector_type(8)))  __bf16 v8bf;
typedef __attribute__((ext_vector_type(8)))  float  v8f;

// ---------------------------------------------------------------------------
// WMMA fragment layouts (wave32, V_WMMA_F32_16X16X32_BF16, ISA 7.12.2):
//  A (16x32 MxK): lane L -> row M=L%16, h=L/16; element e <-> K=16h+e (order used here,
//                 not the ISA table order).  Two contiguous 8-elem chunks.
//  B (32x16 KxN): lane L -> col N=L%16, h=L/16; element e <-> K=16h+e, same order as A.
//                 One contiguous 16-elem chunk.
// ---------------------------------------------------------------------------

__device__ inline v16bf load_a_frag_f32(const float* __restrict__ rowK0, int h) {
  v16bf a;
  const float* p0 = rowK0 + 16 * h;
  const float* p1 = rowK0 + 16 * h + 8;
#pragma unroll
  for (int j = 0; j < 8; ++j) {
    a[j]     = (__bf16)p0[j];
    a[8 + j] = (__bf16)p1[j];
  }
  return a;
}

__device__ inline v16bf load_a_frag_bf(const __bf16* __restrict__ rowK0, int h) {
  v8bf lo = *(const v8bf*)(rowK0 + 16 * h);
  v8bf hi = *(const v8bf*)(rowK0 + 16 * h + 8);
  return __builtin_shufflevector(lo, hi, 0, 1, 2, 3, 4, 5, 6, 7, 8, 9, 10, 11,
                                 12, 13, 14, 15);
}

__device__ inline v16bf load_b_frag_bf(const __bf16* __restrict__ colK0, int h) {
  return *(const v16bf*)(colK0 + 16 * h);
}

// B-fragment from LDS (two ds_load_b128; 16B-aligned, conflict-free layout)
__device__ inline v16bf load_b_frag_lds(const __bf16* p, int h) {
  v8bf lo = *(const v8bf*)(p + 16 * h);
  v8bf hi = *(const v8bf*)(p + 16 * h + 8);
  return __builtin_shufflevector(lo, hi, 0, 1, 2, 3, 4, 5, 6, 7, 8, 9, 10, 11,
                                 12, 13, 14, 15);
}

__device__ inline v8f vzero8() {
  v8f z = {0.f, 0.f, 0.f, 0.f, 0.f, 0.f, 0.f, 0.f};
  return z;
}

__device__ inline void wait_async0() {
#if STAGE_ASYNC
#if __has_builtin(__builtin_amdgcn_s_wait_asynccnt)
  __builtin_amdgcn_s_wait_asynccnt(0);
#else
  asm volatile("s_wait_asynccnt 0x0" ::: "memory");
#endif
#endif
}

// Copy one 16B packet global -> LDS for this lane.
__device__ inline void copy16_to_lds(const __bf16* __restrict__ g, __bf16* l) {
#if STAGE_ASYNC
  unsigned lofs = (unsigned)(size_t)(const void*)l;   // low 32 bits = LDS offset
  asm volatile("global_load_async_to_lds_b128 %0, %1, off"
               :
               : "v"(lofs), "v"(g)
               : "memory");
#else
  *(v8bf*)l = *(const v8bf*)g;
#endif
}

// ---------------------------------------------------------------------------
// Pre-pass: fp32 -> bf16 elementwise (weights; ~0.8M elements total)
// ---------------------------------------------------------------------------
__global__ __launch_bounds__(256) void k_cvt(const float* __restrict__ src,
                                             __bf16* __restrict__ dst, int n) {
  int i = blockIdx.x * blockDim.x + threadIdx.x;
  if (i < n) dst[i] = (__bf16)src[i];
}

// ---------------------------------------------------------------------------
// Kernel 1: z = x @ Wi^T ; gate = silu(x @ Wg^T)
// Block = 256 threads = 8 waves; each wave owns a 16-row token slab
// (128 rows per block). Wi/Wg K-chunks are async-DMA'd into double-buffered
// LDS and shared by all 8 waves: 8x less L2 weight traffic, and staging of
// chunk i+1 overlaps the 64 WMMAs of chunk i. B-fragments are software-
// pipelined one tile ahead so ds_loads stay in flight past each WMMA pair.
// ---------------------------------------------------------------------------
__global__ __launch_bounds__(256) void k_gemm_in(
    const float* __restrict__ x, const __bf16* __restrict__ Wi,
    const __bf16* __restrict__ Wg, float* __restrict__ zbuf,
    __bf16* __restrict__ zbf, float* __restrict__ gbuf) {
  extern __shared__ __bf16 smem[];   // [2][2][128*LDSROW] (buf, matrix, data)

  const int tid = threadIdx.x;
  const int lane = tid & 31;
  const int wave = tid >> 5;                  // 0..7
  const int r = lane & 15, h = lane >> 4;
  const size_t m0 = (size_t)blockIdx.x * 128 + (size_t)wave * 16;
  const float* xrow = x + (m0 + r) * (size_t)DD;

  // Cooperative stage roles: threads 0..127 copy Wi row `tid`,
  // threads 128..255 copy Wg row `tid-128`. One KC-row (256B) per thread.
  const int crow = tid & 127;
  const __bf16* gsrc = (tid < 128 ? Wi : Wg) + (size_t)crow * DD;
  const int ldsoff = (tid < 128 ? 0 : MATELEMS) + crow * LDSROW;

  v8f accz[8], accg[8];
#pragma unroll
  for (int t = 0; t < 8; ++t) { accz[t] = vzero8(); accg[t] = vzero8(); }

  // Prologue: stage chunk 0 into buffer 0.
  {
    __bf16* dst = smem + ldsoff;
#pragma unroll
    for (int c = 0; c < KC / 8; ++c)
      copy16_to_lds(gsrc + c * 8, dst + c * 8);
  }
  wait_async0();
  __syncthreads();

  const int nchunk = DD / KC;   // 16
  int cur = 0;
  for (int ci = 0; ci < nchunk; ++ci) {
    const int kc = ci * KC;

    // Issue async staging of the next chunk into the alternate buffer.
    if (ci + 1 < nchunk) {
      __bf16* dst = smem + (cur ^ 1) * BUFELEMS + ldsoff;
#pragma unroll
      for (int c = 0; c < KC / 8; ++c)
        copy16_to_lds(gsrc + kc + KC + c * 8, dst + c * 8);
    }

    // Compute on the current buffer (B-frags preloaded one tile ahead).
    const __bf16* bWi = smem + cur * BUFELEMS;
    const __bf16* bWg = bWi + MATELEMS;
#pragma unroll
    for (int kk = 0; kk < KC; kk += 32) {
      v16bf a = load_a_frag_f32(xrow + kc + kk, h);
      v16bf nbi = load_b_frag_lds(bWi + r * LDSROW + kk, h);
      v16bf nbg = load_b_frag_lds(bWg + r * LDSROW + kk, h);
#pragma unroll
      for (int t = 0; t < 8; ++t) {
        v16bf bi = nbi, bg = nbg;
        if (t + 1 < 8) {
          nbi = load_b_frag_lds(bWi + ((t + 1) * 16 + r) * LDSROW + kk, h);
          nbg = load_b_frag_lds(bWg + ((t + 1) * 16 + r) * LDSROW + kk, h);
        }
        accz[t] = __builtin_amdgcn_wmma_f32_16x16x32_bf16(
            false, a, false, bi, (short)0, accz[t], false, false);
        accg[t] = __builtin_amdgcn_wmma_f32_16x16x32_bf16(
            false, a, false, bg, (short)0, accg[t], false, false);
      }
    }

    wait_async0();       // next chunk fully in LDS
    __syncthreads();     // all waves done reading current buffer
    cur ^= 1;
  }

#pragma unroll
  for (int t = 0; t < 8; ++t) {
#pragma unroll
    for (int v = 0; v < 8; ++v) {
      size_t m = m0 + v + 8 * h;
      int n = t * 16 + r;
      float zv = accz[t][v];
      zbuf[m * NSCAN + n] = zv;
      zbf[m * NSCAN + n] = (__bf16)zv;
      float g = accg[t][v];
      gbuf[m * NSCAN + n] = g / (1.f + __expf(-g));   // silu
    }
  }
}

// ---------------------------------------------------------------------------
// Kernel 2: dt = softplus(z @ Wdt^T + b_dt); B_in = z @ Wb^T; C_in = z @ Wc^T
// All operands bf16; z-slab read once per wave; weights ~40KB/wave from L2.
// B-fragments preloaded one tile ahead.
// ---------------------------------------------------------------------------
__global__ __launch_bounds__(32) void k_gemm_mid(
    const __bf16* __restrict__ zbf, const __bf16* __restrict__ Wdt,
    const float* __restrict__ b_dt, const __bf16* __restrict__ Wb,
    const __bf16* __restrict__ Wc, float* __restrict__ dtbuf,
    float* __restrict__ Binb, float* __restrict__ Cinb) {
  const int lane = threadIdx.x & 31;
  const int r = lane & 15, h = lane >> 4;
  const size_t m0 = (size_t)blockIdx.x * 16;
  const __bf16* zrow = zbf + (m0 + r) * (size_t)NSCAN;

  v8f accd[8], accb, accc;
#pragma unroll
  for (int t = 0; t < 8; ++t) accd[t] = vzero8();
  accb = vzero8(); accc = vzero8();

#pragma unroll
  for (int k0 = 0; k0 < NSCAN; k0 += 32) {
    v16bf a = load_a_frag_bf(zrow + k0, h);
    v16bf nb = load_b_frag_bf(Wdt + (size_t)r * NSCAN + k0, h);
#pragma unroll
    for (int t = 0; t < 8; ++t) {
      v16bf bd = nb;
      if (t + 1 < 8)
        nb = load_b_frag_bf(Wdt + (size_t)((t + 1) * 16 + r) * NSCAN + k0, h);
      else
        nb = load_b_frag_bf(Wb + (size_t)r * NSCAN + k0, h);
      accd[t] = __builtin_amdgcn_wmma_f32_16x16x32_bf16(
          false, a, false, bd, (short)0, accd[t], false, false);
    }
    v16bf bb = nb;
    v16bf bc = load_b_frag_bf(Wc + (size_t)r * NSCAN + k0, h);
    accb = __builtin_amdgcn_wmma_f32_16x16x32_bf16(
        false, a, false, bb, (short)0, accb, false, false);
    accc = __builtin_amdgcn_wmma_f32_16x16x32_bf16(
        false, a, false, bc, (short)0, accc, false, false);
  }

#pragma unroll
  for (int t = 0; t < 8; ++t) {
    int n = t * 16 + r;
    float bias = b_dt[n];
#pragma unroll
    for (int v = 0; v < 8; ++v) {
      size_t m = m0 + v + 8 * h;
      float p = accd[t][v] + bias;
      float sp = (p > 20.f) ? p : log1pf(__expf(p));   // softplus
      dtbuf[m * NSCAN + n] = sp;
    }
  }
#pragma unroll
  for (int v = 0; v < 8; ++v) {
    size_t m = m0 + v + 8 * h;
    Binb[m * NST + r] = accb[v];
    Cinb[m * NST + r] = accc[v];
  }
}

// ---------------------------------------------------------------------------
// Kernel 3: sequential selective scan over S. One block per batch,
// one thread per scan channel n; 16-wide state in registers.
//   state = (1 + dt*A) * state + (dt*z) * B_t ;  y = <state, C_t> * gate
// Writes y pre-gated as bf16 (feeds output GEMM directly).
// ---------------------------------------------------------------------------
__global__ __launch_bounds__(NSCAN) void k_scan(
    const float* __restrict__ zbuf, const float* __restrict__ gbuf,
    const float* __restrict__ dtbuf, const float* __restrict__ Binb,
    const float* __restrict__ Cinb, const float* __restrict__ A_log,
    __bf16* __restrict__ ybf) {
  const int b = blockIdx.x;
  const int n = threadIdx.x;   // 0..127

  float acoef[NST];
#pragma unroll
  for (int k = 0; k < NST; ++k) acoef[k] = -__expf(A_log[n * NST + k]);

  float st[NST];
#pragma unroll
  for (int k = 0; k < NST; ++k) st[k] = 0.f;

  const size_t base = (size_t)b * SS;
  for (int s = 0; s < SS; ++s) {
    const size_t idx = base + s;
    float dtv = dtbuf[idx * NSCAN + n];
    float zv  = zbuf[idx * NSCAN + n];
    float gv  = gbuf[idx * NSCAN + n];

    float4 B4[4], C4[4];
    const float4* Bp = (const float4*)(Binb + idx * NST);
    const float4* Cp = (const float4*)(Cinb + idx * NST);
#pragma unroll
    for (int q = 0; q < 4; ++q) { B4[q] = Bp[q]; C4[q] = Cp[q]; }
    const float* bv = (const float*)B4;
    const float* cv = (const float*)C4;

    float dz = dtv * zv;
    float y = 0.f;
#pragma unroll
    for (int k = 0; k < NST; ++k) {
      float abar = fmaf(dtv, acoef[k], 1.0f);
      st[k] = fmaf(abar, st[k], dz * bv[k]);
      y = fmaf(st[k], cv[k], y);
    }
    ybf[idx * NSCAN + n] = (__bf16)(y * gv);
  }
}

// ---------------------------------------------------------------------------
// Kernel 4: out = y @ Wo^T.  y bf16 [M,128], Wo bf16 [D,128], out fp32 [M,D].
// Wave computes 64x64 output tile: each B-frag feeds 4 WMMAs.
// Grid (M/64, D/64).
// ---------------------------------------------------------------------------
__global__ __launch_bounds__(32) void k_gemm_out(
    const __bf16* __restrict__ ybf, const __bf16* __restrict__ Wo,
    float* __restrict__ out) {
  const int lane = threadIdx.x & 31;
  const int r = lane & 15, h = lane >> 4;
  const size_t m0 = (size_t)blockIdx.x * 64;
  const int d0 = blockIdx.y * 64;

  v8f acc[4][4];   // [m-subtile][d-subtile]
#pragma unroll
  for (int mi = 0; mi < 4; ++mi)
#pragma unroll
    for (int t = 0; t < 4; ++t) acc[mi][t] = vzero8();

#pragma unroll
  for (int k0 = 0; k0 < NSCAN; k0 += 32) {
    v16bf a[4];
#pragma unroll
    for (int mi = 0; mi < 4; ++mi)
      a[mi] = load_a_frag_bf(ybf + (m0 + mi * 16 + r) * (size_t)NSCAN + k0, h);
    v16bf nb = load_b_frag_bf(Wo + (size_t)(d0 + r) * NSCAN + k0, h);
#pragma unroll
    for (int t = 0; t < 4; ++t) {
      v16bf bw = nb;
      if (t + 1 < 4)
        nb = load_b_frag_bf(Wo + (size_t)(d0 + (t + 1) * 16 + r) * NSCAN + k0, h);
#pragma unroll
      for (int mi = 0; mi < 4; ++mi)
        acc[mi][t] = __builtin_amdgcn_wmma_f32_16x16x32_bf16(
            false, a[mi], false, bw, (short)0, acc[mi][t], false, false);
    }
  }

#pragma unroll
  for (int mi = 0; mi < 4; ++mi)
#pragma unroll
    for (int t = 0; t < 4; ++t)
#pragma unroll
      for (int v = 0; v < 8; ++v) {
        size_t m = m0 + mi * 16 + v + 8 * h;
        out[m * (size_t)DD + d0 + t * 16 + r] = acc[mi][t][v];
      }
}

// ---------------------------------------------------------------------------
extern "C" void kernel_launch(void* const* d_in, const int* in_sizes, int n_in,
                              void* d_out, int out_size, void* d_ws,
                              size_t ws_size, hipStream_t stream) {
  // setup_inputs() order: x, Wi, Wo, A_log, Wb, Wc, Wdt, b_dt, Wg
  const float* x     = (const float*)d_in[0];
  const float* Wi    = (const float*)d_in[1];
  const float* Wo    = (const float*)d_in[2];
  const float* A_log = (const float*)d_in[3];
  const float* Wb    = (const float*)d_in[4];
  const float* Wc    = (const float*)d_in[5];
  const float* Wdt   = (const float*)d_in[6];
  const float* b_dt  = (const float*)d_in[7];
  const float* Wg    = (const float*)d_in[8];
  float* out = (float*)d_out;

  // Workspace partition.
  float* zbuf = (float*)d_ws;                      // [M,128] f32
  float* gbuf = zbuf + (size_t)MM * NSCAN;         // [M,128] f32
  float* dtb  = gbuf + (size_t)MM * NSCAN;         // [M,128] f32
  float* Binb = dtb  + (size_t)MM * NSCAN;         // [M,16]  f32
  float* Cinb = Binb + (size_t)MM * NST;           // [M,16]  f32
  __bf16* zbf   = (__bf16*)(Cinb + (size_t)MM * NST);  // [M,128] bf16
  __bf16* ybf   = zbf + (size_t)MM * NSCAN;            // [M,128] bf16
  __bf16* Wi_bf = ybf + (size_t)MM * NSCAN;            // [128,2048]
  __bf16* Wg_bf = Wi_bf + (size_t)NSCAN * DD;          // [128,2048]
  __bf16* Wdt_bf = Wg_bf + (size_t)NSCAN * DD;         // [128,128]
  __bf16* Wb_bf  = Wdt_bf + (size_t)NSCAN * NSCAN;     // [16,128]
  __bf16* Wc_bf  = Wb_bf + (size_t)NST * NSCAN;        // [16,128]
  __bf16* Wo_bf  = Wc_bf + (size_t)NST * NSCAN;        // [2048,128]

  dim3 b256(256, 1, 1);
  dim3 w32(32, 1, 1);

  // Weight pre-conversion to bf16 (tiny; L2-resident afterwards).
  k_cvt<<<dim3((NSCAN * DD + 255) / 256), b256, 0, stream>>>(Wi, Wi_bf, NSCAN * DD);
  k_cvt<<<dim3((NSCAN * DD + 255) / 256), b256, 0, stream>>>(Wg, Wg_bf, NSCAN * DD);
  k_cvt<<<dim3((NSCAN * NSCAN + 255) / 256), b256, 0, stream>>>(Wdt, Wdt_bf, NSCAN * NSCAN);
  k_cvt<<<dim3((NST * NSCAN + 255) / 256), b256, 0, stream>>>(Wb, Wb_bf, NST * NSCAN);
  k_cvt<<<dim3((NST * NSCAN + 255) / 256), b256, 0, stream>>>(Wc, Wc_bf, NST * NSCAN);
  k_cvt<<<dim3((DD * NSCAN + 255) / 256), b256, 0, stream>>>(Wo, Wo_bf, DD * NSCAN);

  k_gemm_in<<<dim3(MM / 128, 1, 1), b256, SHBYTES, stream>>>(x, Wi_bf, Wg_bf,
                                                             zbuf, zbf, gbuf);
  k_gemm_mid<<<dim3(MM / 16, 1, 1), w32, 0, stream>>>(zbf, Wdt_bf, b_dt, Wb_bf,
                                                      Wc_bf, dtb, Binb, Cinb);
  k_scan<<<dim3(BB, 1, 1), dim3(NSCAN, 1, 1), 0, stream>>>(zbuf, gbuf, dtb,
                                                           Binb, Cinb, A_log,
                                                           ybf);
  k_gemm_out<<<dim3(MM / 64, DD / 64, 1), w32, 0, stream>>>(ybf, Wo_bf, out);
}
